// TimmFace_28269474742388
// MI455X (gfx1250) — matrix operand order repaired and checked by reading, not compiled, not run on hardware
//
#include <hip/hip_runtime.h>
#include <hip/hip_bf16.h>

typedef __attribute__((ext_vector_type(16))) _Float16 v16h;
typedef __attribute__((ext_vector_type(8)))  _Float16 v8h;
typedef __attribute__((ext_vector_type(8)))  float    v8f;

#define BATCH   1024
#define EDIM    512
#define NCLS    262144
#define NSAMP   16384
#define MARGIN  0.5f
#define SCALE   64.0f
#define BN_EPS  1e-5f
#define NCHUNK  16
#define CW      1024   // columns per chunk
#define MT      64     // M tiles (BATCH/16)

// ---------------- utility -----------------
__global__ void zero_kernel(int* p, int n) {
    int i = blockIdx.x * 256 + threadIdx.x;
    if (i < n) p[i] = 0;
}

// ---------------- BatchNorm stats ----------------
__global__ __launch_bounds__(256)
void bn_stats_kernel(const float* __restrict__ embs, float* __restrict__ mu,
                     float* __restrict__ rinv) {
    int c = blockIdx.x * 256 + threadIdx.x;   // 512 columns total
    float s = 0.f, ss = 0.f;
    for (int b = 0; b < BATCH; ++b) {
        float v = embs[(size_t)b * EDIM + c];
        s += v; ss += v * v;
    }
    float m   = s * (1.0f / BATCH);
    float var = ss * (1.0f / BATCH) - m * m;
    mu[c]   = m;
    rinv[c] = rsqrtf(var + BN_EPS);
}

// ---------------- row normalize -> f16 ----------------
__global__ __launch_bounds__(256)
void rownorm_kernel(const float* __restrict__ embs, const float* __restrict__ mu,
                    const float* __restrict__ rinv, _Float16* __restrict__ xn) {
    __shared__ float red[256];
    int r = blockIdx.x, tid = threadIdx.x;
    float x0 = (embs[(size_t)r * EDIM + tid]       - mu[tid])       * rinv[tid];
    float x1 = (embs[(size_t)r * EDIM + 256 + tid] - mu[256 + tid]) * rinv[256 + tid];
    red[tid] = x0 * x0 + x1 * x1;
    __syncthreads();
    for (int s = 128; s > 0; s >>= 1) {
        if (tid < s) red[tid] += red[tid + s];
        __syncthreads();
    }
    float inv = rsqrtf(red[0]);
    xn[(size_t)r * EDIM + tid]       = (_Float16)(x0 * inv);
    xn[(size_t)r * EDIM + 256 + tid] = (_Float16)(x1 * inv);
}

// ---------------- selection (exact top-NS via 2-pass radix histogram) ----------------
__global__ void mark_kernel(const int* __restrict__ labels, int* __restrict__ mask) {
    int i = blockIdx.x * 256 + threadIdx.x;   // 1024
    mask[labels[i]] = 1;
}

__device__ __forceinline__ unsigned int key_of(const float* perm, const int* mask, int c) {
    return mask[c] ? 0x40000000u : __float_as_uint(perm[c]);
}

__global__ __launch_bounds__(256)
void hist1_kernel(const float* __restrict__ perm, const int* __restrict__ mask,
                  unsigned int* __restrict__ hist) {
    int c = blockIdx.x * 256 + threadIdx.x;   // NCLS
    unsigned int k = key_of(perm, mask, c);
    atomicAdd(&hist[k >> 16], 1u);
}

__global__ __launch_bounds__(256)
void select1_kernel(const unsigned int* __restrict__ hist, int* __restrict__ sel) {
    __shared__ unsigned int cs[256];
    int tid = threadIdx.x;
    unsigned int s = 0;
    for (int j = 0; j < 256; ++j) s += hist[tid * 256 + j];
    cs[tid] = s;
    __syncthreads();
    if (tid == 0) {
        unsigned int acc = 0; int chunkI = 0;
        for (int i = 255; i >= 0; --i) {
            if (acc + cs[i] >= (unsigned)NSAMP) { chunkI = i; break; }
            acc += cs[i];
        }
        int tb = 0;
        for (int b = 255; b >= 0; --b) {
            int bin = chunkI * 256 + b;
            if (acc + hist[bin] >= (unsigned)NSAMP) { tb = bin; break; }
            acc += hist[bin];
        }
        sel[0] = tb;                       // high-16 threshold bin
        sel[1] = NSAMP - (int)acc;         // still needed inside bin
    }
}

__global__ __launch_bounds__(256)
void hist2_kernel(const float* __restrict__ perm, const int* __restrict__ mask,
                  const int* __restrict__ sel, unsigned int* __restrict__ hist) {
    int c = blockIdx.x * 256 + threadIdx.x;
    unsigned int tb = (unsigned)sel[0];
    unsigned int k = key_of(perm, mask, c);
    if ((k >> 16) == tb) atomicAdd(&hist[k & 0xFFFFu], 1u);
}

__global__ __launch_bounds__(256)
void select2_kernel(const unsigned int* __restrict__ hist, int* __restrict__ sel) {
    __shared__ unsigned int cs[256];
    int tid = threadIdx.x;
    unsigned int s = 0;
    for (int j = 0; j < 256; ++j) s += hist[tid * 256 + j];
    cs[tid] = s;
    __syncthreads();
    if (tid == 0) {
        unsigned int need = (unsigned)sel[1];
        unsigned int acc = 0; int chunkI = 0;
        for (int i = 255; i >= 0; --i) {
            if (acc + cs[i] >= need) { chunkI = i; break; }
            acc += cs[i];
        }
        int lb = 0;
        for (int b = 255; b >= 0; --b) {
            int bin = chunkI * 256 + b;
            if (acc + hist[bin] >= need) { lb = bin; break; }
            acc += hist[bin];
        }
        sel[2] = (sel[0] << 16) | lb;      // exact 32-bit threshold key
        sel[3] = (int)(need - acc);        // # ties (== key) to keep, lowest class ids
    }
}

// ---- 3-phase prefix scan over (key>T, key==T) flags: rank + sorted idx ----
__global__ __launch_bounds__(256)
void scanA_kernel(const float* __restrict__ perm, const int* __restrict__ mask,
                  const int* __restrict__ sel, int* __restrict__ totG,
                  int* __restrict__ totT) {
    __shared__ int rg[256], rt[256];
    int tid = threadIdx.x;
    int base = blockIdx.x * 1024 + tid * 4;
    unsigned int T = (unsigned)sel[2];
    int g = 0, t = 0;
    for (int j = 0; j < 4; ++j) {
        unsigned int k = key_of(perm, mask, base + j);
        g += (k > T); t += (k == T);
    }
    rg[tid] = g; rt[tid] = t;
    __syncthreads();
    for (int s = 128; s > 0; s >>= 1) {
        if (tid < s) { rg[tid] += rg[tid + s]; rt[tid] += rt[tid + s]; }
        __syncthreads();
    }
    if (tid == 0) { totG[blockIdx.x] = rg[0]; totT[blockIdx.x] = rt[0]; }
}

__global__ __launch_bounds__(256)
void scanB_kernel(const int* __restrict__ totG, const int* __restrict__ totT,
                  int* __restrict__ offG, int* __restrict__ offT) {
    __shared__ int sg[256], st[256];
    int tid = threadIdx.x;
    sg[tid] = totG[tid]; st[tid] = totT[tid];
    __syncthreads();
    for (int d = 1; d < 256; d <<= 1) {
        int vg = (tid >= d) ? sg[tid - d] : 0;
        int vt = (tid >= d) ? st[tid - d] : 0;
        __syncthreads();
        sg[tid] += vg; st[tid] += vt;
        __syncthreads();
    }
    offG[tid] = sg[tid] - totG[tid];   // exclusive
    offT[tid] = st[tid] - totT[tid];
}

__global__ __launch_bounds__(256)
void scanC_kernel(const float* __restrict__ perm, const int* __restrict__ mask,
                  const int* __restrict__ sel, const int* __restrict__ offG,
                  const int* __restrict__ offT, int* __restrict__ rank,
                  int* __restrict__ idx) {
    __shared__ int rg[256], rt[256];
    int tid = threadIdx.x;
    int base = blockIdx.x * 1024 + tid * 4;
    unsigned int T = (unsigned)sel[2];
    int e = sel[3];
    int g4[4], t4[4];
    int g = 0, t = 0;
    for (int j = 0; j < 4; ++j) {
        unsigned int k = key_of(perm, mask, base + j);
        g4[j] = (k > T); t4[j] = (k == T);
        g += g4[j]; t += t4[j];
    }
    rg[tid] = g; rt[tid] = t;
    __syncthreads();
    for (int d = 1; d < 256; d <<= 1) {
        int vg = (tid >= d) ? rg[tid - d] : 0;
        int vt = (tid >= d) ? rt[tid - d] : 0;
        __syncthreads();
        rg[tid] += vg; rt[tid] += vt;
        __syncthreads();
    }
    int exG = offG[blockIdx.x] + rg[tid] - g;   // exclusive before this thread
    int exT = offT[blockIdx.x] + rt[tid] - t;
    for (int j = 0; j < 4; ++j) {
        int c = base + j;
        int r = exG + (exT < e ? exT : e);      // rank among selected, sorted order
        rank[c] = r;
        int selc = g4[j] | (t4[j] & (exT < e ? 1 : 0));
        if (selc) idx[r] = c;
        exG += g4[j]; exT += t4[j];
    }
}

__global__ void newlab_kernel(const int* __restrict__ labels, const int* __restrict__ rank,
                              int* __restrict__ nl) {
    int i = blockIdx.x * 256 + threadIdx.x;   // 1024
    nl[i] = rank[labels[i]];                  // labels always selected (key==2.0 bits > T)
}

// ---------------- gather + normalize weights -> f16 ----------------
__global__ __launch_bounds__(256)
void gather_kernel(const float* __restrict__ weight, const int* __restrict__ idx,
                   _Float16* __restrict__ wn) {
    __shared__ float red[256];
    int n = blockIdx.x, tid = threadIdx.x;
    int r = idx[n];
    const float* src = weight + (size_t)r * EDIM;
    float w0 = src[tid], w1 = src[256 + tid];
    red[tid] = w0 * w0 + w1 * w1;
    __syncthreads();
    for (int s = 128; s > 0; s >>= 1) {
        if (tid < s) red[tid] += red[tid + s];
        __syncthreads();
    }
    float inv = rsqrtf(red[0]);
    wn[(size_t)n * EDIM + tid]       = (_Float16)(w0 * inv);
    wn[(size_t)n * EDIM + 256 + tid] = (_Float16)(w1 * inv);
}

// ---------------- fused WMMA GEMM + ArcFace + streaming softmax ----------------
__global__ __launch_bounds__(256)
void gemm_softmax_kernel(const _Float16* __restrict__ xn, const _Float16* __restrict__ wn,
                         const int* __restrict__ nlab, float* __restrict__ pmax,
                         float* __restrict__ psum, float* __restrict__ pll) {
    __shared__ _Float16 As[16 * EDIM];      // 16 KB A tile
    __shared__ int labS[16];
    __shared__ float sMax[8][16], sSum[8][16], sLL[8][16];

    const int mtile = blockIdx.x;
    const int chunk = blockIdx.y;
    const int tid = threadIdx.x;

    {   // A tile: 16 consecutive rows of xn -> LDS (contiguous 16 KB)
        const uint4* src = (const uint4*)(xn + (size_t)mtile * 16 * EDIM);
        uint4* dst = (uint4*)As;
        #pragma unroll
        for (int j = 0; j < 4; ++j) dst[tid + 256 * j] = src[tid + 256 * j];
    }
    if (tid < 16) labS[tid] = nlab[mtile * 16 + tid];
    __syncthreads();

    const int wave  = tid >> 5;
    const int lane  = tid & 31;
    const int lhalf = lane >> 4;   // 0|1
    const int l15   = lane & 15;

    int labr[8];
    #pragma unroll
    for (int r = 0; r < 8; ++r) labr[r] = labS[r + lhalf * 8];

    float rmax[8], rsum[8], rll[8];
    #pragma unroll
    for (int r = 0; r < 8; ++r) { rmax[r] = -1e30f; rsum[r] = 0.f; rll[r] = 0.f; }

    // A fragments for all 16 K-steps, hoisted into registers once (128 VGPRs).
    // A uses contiguous k order (B matches): lanes0-15 K{0..15}, lanes16-31 K{16..31}.
    const _Float16* aBase = As + (size_t)l15 * EDIM + lhalf * 16;
    v16h afrag[16];
    #pragma unroll
    for (int ks = 0; ks < 16; ++ks) {
        v8h alo = *(const v8h*)(aBase + ks * 32);
        v8h ahi = *(const v8h*)(aBase + ks * 32 + 8);
        afrag[ks] = __builtin_shufflevector(alo, ahi,
                     0,1,2,3,4,5,6,7,8,9,10,11,12,13,14,15);
    }

    for (int t = wave; t < CW / 16; t += 8) {
        const int nbase = chunk * CW + t * 16;
        const int ncol  = nbase + l15;
        // B fragment: lane = column, lanes0-15 K{0..15}, lanes16-31 K{16..31} (contiguous 32B)
        const _Float16* bBase = wn + (size_t)ncol * EDIM + lhalf * 16;

        // prefetch next tile's B row (t+8 -> +128 columns); separate BB, does not
        // interfere with the sched_group_barrier groups below
        if (t + 8 < CW / 16) {
            const char* pf = (const char*)(wn + (size_t)(ncol + 128) * EDIM);
            #pragma unroll
            for (int pb = 0; pb < 8; ++pb) __builtin_prefetch(pf + pb * 128, 0, 3);
        }

        v8f acc = {0.f, 0.f, 0.f, 0.f, 0.f, 0.f, 0.f, 0.f};
        #pragma unroll
        for (int ks = 0; ks < 16; ++ks) {
            v16h b = *(const v16h*)(bBase + ks * 32);
            acc = __builtin_amdgcn_wmma_f32_16x16x32_f16(false, afrag[ks], false, b,
                                                         (short)0, acc, false, false);
        }
        // Software pipeline directive: 4-fragment-deep B prefetch.
        //   8 VMEM reads (b128) up front, then steady state {1 WMMA, 2 VMEM reads},
        //   then drain 4 WMMAs.  mask 0x20 = VMEM read, 0x8 = MFMA/WMMA.
        __builtin_amdgcn_sched_group_barrier(0x020, 8, 0);
        #pragma unroll
        for (int ks = 0; ks < 12; ++ks) {
            __builtin_amdgcn_sched_group_barrier(0x008, 1, 0);
            __builtin_amdgcn_sched_group_barrier(0x020, 2, 0);
        }
        __builtin_amdgcn_sched_group_barrier(0x008, 4, 0);

        // epilogue: C layout VGPR r, lanes0-15 -> row r col l15 ; lanes16-31 -> row 8+r
        #pragma unroll
        for (int r = 0; r < 8; ++r) {
            float v = acc[r];
            v = fminf(fmaxf(v, -0.999f), 0.999f);        // reference clips ALL logits
            bool ismatch = (ncol == labr[r]);
            if (ismatch) v = __cosf(acosf(v) + MARGIN);  // ArcFace margin at label
            v *= SCALE;
            if (ismatch) rll[r] += v;
            if (v > rmax[r]) { rsum[r] = rsum[r] * __expf(rmax[r] - v) + 1.0f; rmax[r] = v; }
            else             { rsum[r] += __expf(v - rmax[r]); }
        }
    }

    // merge across the 16 lanes that share each row (xor within 16-lane group)
    #pragma unroll
    for (int r = 0; r < 8; ++r) {
        #pragma unroll
        for (int m = 1; m <= 8; m <<= 1) {
            float om = __shfl_xor(rmax[r], m, 32);
            float os = __shfl_xor(rsum[r], m, 32);
            float ol = __shfl_xor(rll[r],  m, 32);
            float nm = fmaxf(rmax[r], om);
            rsum[r] = rsum[r] * __expf(rmax[r] - nm) + os * __expf(om - nm);
            rmax[r] = nm;
            rll[r] += ol;
        }
    }
    if (l15 == 0) {
        #pragma unroll
        for (int r = 0; r < 8; ++r) {
            int row = r + lhalf * 8;
            sMax[wave][row] = rmax[r];
            sSum[wave][row] = rsum[r];
            sLL[wave][row]  = rll[r];
        }
    }
    __syncthreads();
    if (tid < 16) {
        float gm = -1e30f, gs = 0.f, gl = 0.f;
        for (int w = 0; w < 8; ++w) {
            float om = sMax[w][tid], os = sSum[w][tid];
            float nm = fmaxf(gm, om);
            gs = gs * __expf(gm - nm) + os * __expf(om - nm);
            gm = nm;
            gl += sLL[w][tid];
        }
        int p = (chunk * MT + mtile) * 16 + tid;
        pmax[p] = gm; psum[p] = gs; pll[p] = gl;
    }
}

// ---------------- final combine + mean CE loss ----------------
__global__ __launch_bounds__(1024)
void combine_kernel(const float* __restrict__ pmax, const float* __restrict__ psum,
                    const float* __restrict__ pll, float* __restrict__ out) {
    __shared__ float red[1024];
    int row = threadIdx.x;
    int mtile = row >> 4, rl = row & 15;
    float gm = -1e30f, gs = 0.f, gl = 0.f;
    for (int c = 0; c < NCHUNK; ++c) {
        int p = (c * MT + mtile) * 16 + rl;
        float om = pmax[p], os = psum[p];
        float nm = fmaxf(gm, om);
        gs = gs * __expf(gm - nm) + os * __expf(om - nm);
        gm = nm;
        gl += pll[p];
    }
    red[row] = (gm + logf(gs)) - gl;   // logsumexp - label logit
    __syncthreads();
    for (int s = 512; s > 0; s >>= 1) {
        if (row < s) red[row] += red[row + s];
        __syncthreads();
    }
    if (row == 0) out[0] = red[0] / (float)BATCH;
}

// ---------------- host launcher ----------------
extern "C" void kernel_launch(void* const* d_in, const int* in_sizes, int n_in,
                              void* d_out, int out_size, void* d_ws, size_t ws_size,
                              hipStream_t stream) {
    const float* embs   = (const float*)d_in[0];   // [1024,512]
    const float* weight = (const float*)d_in[1];   // [262144,512]
    const float* perm   = (const float*)d_in[2];   // [262144]
    const int*   labels = (const int*)d_in[3];     // [1024]
    float* out = (float*)d_out;

    char* w = (char*)d_ws;
    size_t off = 0;
    auto alloc = [&](size_t bytes) -> char* {
        char* p = w + off;
        off = (off + bytes + 255) & ~(size_t)255;
        return p;
    };
    float*        mu    = (float*)alloc(EDIM * 4);
    float*        rinv  = (float*)alloc(EDIM * 4);
    _Float16*     xn    = (_Float16*)alloc((size_t)BATCH * EDIM * 2);
    _Float16*     wn    = (_Float16*)alloc((size_t)NSAMP * EDIM * 2);
    int*          mask  = (int*)alloc((size_t)NCLS * 4);
    unsigned int* hist1 = (unsigned int*)alloc(65536 * 4);
    unsigned int* hist2 = (unsigned int*)alloc(65536 * 4);
    int*          sel   = (int*)alloc(64);
    int*          totG  = (int*)alloc(256 * 4);
    int*          totT  = (int*)alloc(256 * 4);
    int*          offG  = (int*)alloc(256 * 4);
    int*          offT  = (int*)alloc(256 * 4);
    int*          rank  = (int*)alloc((size_t)NCLS * 4);
    int*          idx   = (int*)alloc((size_t)NSAMP * 4);
    int*          nl    = (int*)alloc(BATCH * 4);
    float*        pmax  = (float*)alloc((size_t)NCHUNK * MT * 16 * 4);
    float*        psum  = (float*)alloc((size_t)NCHUNK * MT * 16 * 4);
    float*        pll   = (float*)alloc((size_t)NCHUNK * MT * 16 * 4);
    (void)ws_size; (void)n_in; (void)in_sizes; (void)out_size;

    zero_kernel<<<(NCLS + 255) / 256, 256, 0, stream>>>(mask, NCLS);
    zero_kernel<<<(65536 + 255) / 256, 256, 0, stream>>>((int*)hist1, 65536);
    zero_kernel<<<(65536 + 255) / 256, 256, 0, stream>>>((int*)hist2, 65536);

    bn_stats_kernel<<<2, 256, 0, stream>>>(embs, mu, rinv);
    rownorm_kernel<<<BATCH, 256, 0, stream>>>(embs, mu, rinv, xn);

    mark_kernel<<<4, 256, 0, stream>>>(labels, mask);
    hist1_kernel<<<NCLS / 256, 256, 0, stream>>>(perm, mask, hist1);
    select1_kernel<<<1, 256, 0, stream>>>(hist1, sel);
    hist2_kernel<<<NCLS / 256, 256, 0, stream>>>(perm, mask, sel, hist2);
    select2_kernel<<<1, 256, 0, stream>>>(hist2, sel);

    scanA_kernel<<<256, 256, 0, stream>>>(perm, mask, sel, totG, totT);
    scanB_kernel<<<1, 256, 0, stream>>>(totG, totT, offG, offT);
    scanC_kernel<<<256, 256, 0, stream>>>(perm, mask, sel, offG, offT, rank, idx);
    newlab_kernel<<<4, 256, 0, stream>>>(labels, rank, nl);

    gather_kernel<<<NSAMP, 256, 0, stream>>>(weight, idx, wn);

    gemm_softmax_kernel<<<dim3(MT, NCHUNK), 256, 0, stream>>>(xn, wn, nl, pmax, psum, pll);
    combine_kernel<<<1, 1024, 0, stream>>>(pmax, psum, pll, out);
}
